// QuantumKernelAttentionFidelity_57690000720068
// MI455X (gfx1250) — compile-verified, matrix operand order checked by reading, not run on hardware
//
#include <hip/hip_runtime.h>
#include <hip/hip_bf16.h>

typedef _Float16 f16;
typedef __attribute__((ext_vector_type(16))) _Float16 v16h;
typedef __attribute__((ext_vector_type(8)))  float    v8f;

#define NB 8
#define NH 8
#define NL 1024    // L == S == 1024
#define ND 64
#define PI_F 3.14159265358979f
#define CT_STRIDE 136   // padded LDS row pitch (f16): 272B, keeps b128 frags aligned + bank-spread

union Frag {
    uint4 q[2];
    v16h  h;
};

// Load one 16x32 f16 A/B fragment whose "rows" (M for A, N for B) are 32-element
// runs with pitch strideF16.  CDNA5 16-bit layout: lane (m, hi) needs
// K = hi*8 + {0..7} and 16 + hi*8 + {0..7}  ->  exactly two b128 loads.
__device__ inline v16h load_frag_s(const f16* tile, int lane, int strideF16) {
    int r  = lane & 15;
    int hi = lane >> 4;
    const uint4* p = (const uint4*)(tile + (size_t)r * strideF16);
    Frag f;
    f.q[0] = p[hi];      // K = hi*8 .. hi*8+7
    f.q[1] = p[2 + hi];  // K = 16+hi*8 .. 16+hi*8+7
    return f.h;
}

__device__ inline v8f wmma_f16(v16h a, v16h b, v8f c) {
    return __builtin_amdgcn_wmma_f32_16x16x32_f16(false, a, false, b, (short)0, c, false, false);
}

// ---------------------------------------------------------------------------
// 1) Encode: one wave per token row. Angles then the closed-form 2-layer
//    ZZ-feature-map state, stored as f16 rows of 32 = [Re(0..15)|Im(0..15)].
//    For Q also stores the variant [Im | -Re] used for fid.im.
// ---------------------------------------------------------------------------
__global__ void qka_encode(const float* __restrict__ x,   // [B, LS, H, 64]
                           const float* __restrict__ w,   // [4, 64]
                           const float* __restrict__ bias,// [4]
                           f16* __restrict__ SA,          // [B,H,LS,32]  [Re|Im]
                           f16* __restrict__ SB,          // [B,H,LS,32]  [Im|-Re]
                           int isq) {
    int wave = threadIdx.x >> 5;
    int lane = threadIdx.x & 31;
    int row  = blockIdx.x * 8 + wave;          // [0, B*H*LS)
    int b    = row / (NH * NL);
    int rem  = row - b * (NH * NL);
    int h    = rem / NL;
    int l    = rem - h * NL;

    const float* xr = x + (((size_t)(b * NL + l)) * NH + h) * ND;
    float x0 = xr[lane];
    float x1 = xr[lane + 32];
    float p0 = x0 * w[lane      ] + x1 * w[lane + 32 ];
    float p1 = x0 * w[lane + 64 ] + x1 * w[lane + 96 ];
    float p2 = x0 * w[lane + 128] + x1 * w[lane + 160];
    float p3 = x0 * w[lane + 192] + x1 * w[lane + 224];
#pragma unroll
    for (int m = 16; m >= 1; m >>= 1) {
        p0 += __shfl_xor(p0, m);
        p1 += __shfl_xor(p1, m);
        p2 += __shfl_xor(p2, m);
        p3 += __shfl_xor(p3, m);
    }
    float a0 = tanhf(p0 + bias[0]) * PI_F;
    float a1 = tanhf(p1 + bias[1]) * PI_F;
    float a2 = tanhf(p2 + bias[2]) * PI_F;
    float a3 = tanhf(p3 + bias[3]) * PI_F;

    int d = lane & 15;                 // basis state (lanes 16..31 mirror 0..15)
    float z0 = (d & 8) ? -1.f : 1.f;   // wire 0 is MSB
    float z1 = (d & 4) ? -1.f : 1.f;
    float z2 = (d & 2) ? -1.f : 1.f;
    float z3 = (d & 1) ? -1.f : 1.f;
    float arg = a0 * z0 + a1 * z1 + a2 * z2 + a3 * z3
              + a0 * a1 * z0 * z1 + a1 * a2 * z1 * z2 + a2 * a3 * z2 * z3;
    float c = cosf(0.5f * arg);
    float s = sinf(0.5f * arg);
    float dre = c, dim = -s;           // exp(-i*arg/2)

    // S(d) = sum_j (-1)^popc(j&d) * diag[j]   via lane broadcasts
    float Sre = 0.f, Sim = 0.f;
#pragma unroll
    for (int j = 0; j < 16; ++j) {
        float jr = __shfl(dre, j);
        float ji = __shfl(dim, j);
        float sg = (__popc(j & d) & 1) ? -1.f : 1.f;
        Sre += sg * jr;
        Sim += sg * ji;
    }
    float str = 0.0625f * (dre * Sre - dim * Sim);
    float sti = 0.0625f * (dre * Sim + dim * Sre);

    if (lane < 16) {
        f16* ra = SA + (size_t)row * 32;
        ra[d]      = (f16)str;
        ra[16 + d] = (f16)sti;
        if (isq) {
            f16* rb = SB + (size_t)row * 32;
            rb[d]      = (f16)sti;
            rb[16 + d] = (f16)(-str);
        }
    }
}

// ---------------------------------------------------------------------------
// 2) V layout change: [B,S,H,64] f32 -> [B,H,S,64] f16
// ---------------------------------------------------------------------------
__global__ void qka_convert_v(const float* __restrict__ v, f16* __restrict__ vh) {
    size_t i = (size_t)blockIdx.x * 256 + threadIdx.x;   // < B*H*S*64
    int d = (int)(i & 63);
    size_t t = i >> 6;
    int s = (int)(t & (NL - 1));
    t >>= 10;
    int h = (int)(t & 7);
    int b = (int)(t >> 3);
    vh[i] = (f16)v[(((size_t)b * NL + s) * NH + h) * ND + d];
}

// ---------------------------------------------------------------------------
// 3) Scores + fused normalize: wave owns a 16-row L-tile.
//    Sweep 1: fid.re/fid.im via two v_wmma_f32_16x16x32_f16 per 16x16 tile,
//             score = |fid|^2 -> attn (unnormalized), accumulate row sums.
//    Sweep 2: after s_wait_storecnt 0, rescale own rows in place (L0/L2 hits),
//             whole wave per row with float4 for 512B/instruction.
// ---------------------------------------------------------------------------
__global__ void qka_scores(const f16* __restrict__ QA,
                           const f16* __restrict__ QB,
                           const f16* __restrict__ KS,
                           float* __restrict__ attn) {   // [B,H,L,S]
    int wave  = threadIdx.x >> 5;
    int lane  = threadIdx.x & 31;
    int gtile = blockIdx.x * 8 + wave;        // 0 .. B*H*(L/16)-1
    int bh    = gtile >> 6;                   // L/16 == 64 tiles per (b,h)
    int lt    = gtile & 63;

    v16h are = load_frag_s(QA + ((size_t)bh * NL + lt * 16) * 32, lane, 32);
    v16h aim = load_frag_s(QB + ((size_t)bh * NL + lt * 16) * 32, lane, 32);
    const f16* ks = KS + (size_t)bh * NL * 32;
    float* arow = attn + (size_t)bh * NL * NL + (size_t)(lt * 16) * NL;

    int n  = lane & 15;
    int hi = lane >> 4;
    float rs[8] = {0.f, 0.f, 0.f, 0.f, 0.f, 0.f, 0.f, 0.f};

    for (int st = 0; st < 64; ++st) {
        v16h bf = load_frag_s(ks + (size_t)(st * 16) * 32, lane, 32);
        v8f cre = {};
        v8f cim = {};
        cre = wmma_f16(are, bf, cre);
        cim = wmma_f16(aim, bf, cim);
#pragma unroll
        for (int r = 0; r < 8; ++r) {
            float sc = cre[r] * cre[r] + cim[r] * cim[r];
            rs[r] += sc;
            arow[(size_t)(r + 8 * hi) * NL + st * 16 + n] = sc;
        }
    }
    // reduce the 16 columns held by each half-wave (D layout: lanes 0-15 row r,
    // lanes 16-31 row r+8); after this every lane holds its row's full sum.
#pragma unroll
    for (int r = 0; r < 8; ++r) {
        float v = rs[r];
        v += __shfl_xor(v, 1);
        v += __shfl_xor(v, 2);
        v += __shfl_xor(v, 4);
        v += __shfl_xor(v, 8);
        rs[r] = 1.f / (v + 1e-6f);
    }

    // make sweep-1 stores visible before cross-lane re-read
    asm volatile("s_wait_storecnt 0x0" ::: "memory");

    // sweep 2: normalize own 16 rows in place, coalesced
#pragma unroll 1
    for (int m = 0; m < 16; ++m) {
        float inv = __shfl(rs[m & 7], (m >> 3) << 4);
        float4* prow = (float4*)(arow + (size_t)m * NL);
#pragma unroll
        for (int i = 0; i < 8; ++i) {
            float4 v = prow[lane + i * 32];
            v.x *= inv; v.y *= inv; v.z *= inv; v.w *= inv;
            prow[lane + i * 32] = v;
        }
    }
}

// ---------------------------------------------------------------------------
// 4) Context: C[L,64] = W[L,S] @ V[S,64] per (b,h).  Block = 32 L-rows x 64 D,
//    8 waves = 2 l-tiles x 4 d-tiles.  K=128 per stage (4 WMMAs/wave/stage,
//    8 stages) via LDS tiles with padded pitch CT_STRIDE.
// ---------------------------------------------------------------------------
__global__ void qka_context(const float* __restrict__ attn,  // [B,H,L,S] normalized
                            const f16* __restrict__ vh,      // [B,H,S,64]
                            float* __restrict__ out) {       // [B,L,H,64]
    __shared__ f16 Wt[32 * CT_STRIDE];   // [row][k]
    __shared__ f16 Vt[64 * CT_STRIDE];   // [d][k]  (transposed: B frags contiguous)

    int bh   = blockIdx.x >> 5;   // L/32 == 32 blocks per (b,h)
    int lblk = blockIdx.x & 31;
    int b = bh >> 3, h = bh & 7;
    int wave = threadIdx.x >> 5;
    int lane = threadIdx.x & 31;
    int lt = wave >> 2;           // 0..1
    int dt = wave & 3;            // 0..3

    const float* wbase = attn + (size_t)bh * NL * NL + (size_t)(lblk * 32) * NL;
    const f16*   vbase = vh + (size_t)bh * NL * ND;

    v8f acc = {};
    for (int s0 = 0; s0 < NL; s0 += 128) {
        __syncthreads();
        {   // stage W: 32 rows x 128 cols f32 -> f16 (16 floats / thread)
            int t   = threadIdx.x;
            int row = t >> 3;
            int c0  = (t & 7) << 4;
            const float* src = wbase + (size_t)row * NL + s0 + c0;
            f16* dst = Wt + row * CT_STRIDE + c0;
#pragma unroll
            for (int i = 0; i < 4; ++i) {
                float4 wv = *(const float4*)(src + 4 * i);
                dst[4 * i + 0] = (f16)wv.x;
                dst[4 * i + 1] = (f16)wv.y;
                dst[4 * i + 2] = (f16)wv.z;
                dst[4 * i + 3] = (f16)wv.w;
            }
        }
        {   // stage V transposed: Vt[d][k] = vh[s0+k][d]; u32 global reads (coalesced)
#pragma unroll
            for (int i = 0; i < 16; ++i) {
                int idx = threadIdx.x + i * 256;   // 0..4095
                int k   = idx >> 5;                // 0..127
                int d2  = (idx & 31) << 1;         // 0..62
                const f16* sp = vbase + (size_t)(s0 + k) * ND + d2;
                Vt[d2 * CT_STRIDE + k]       = sp[0];
                Vt[(d2 + 1) * CT_STRIDE + k] = sp[1];
            }
        }
        __syncthreads();
#pragma unroll
        for (int kk = 0; kk < 4; ++kk) {
            v16h af = load_frag_s(Wt + lt * 16 * CT_STRIDE + kk * 32, lane, CT_STRIDE);
            v16h bf = load_frag_s(Vt + dt * 16 * CT_STRIDE + kk * 32, lane, CT_STRIDE);
            acc = wmma_f16(af, bf, acc);
        }
    }

    int n  = lane & 15;
    int hi = lane >> 4;
    int l0 = lblk * 32 + lt * 16;
#pragma unroll
    for (int r = 0; r < 8; ++r) {
        int l = l0 + r + 8 * hi;
        out[(((size_t)b * NL + l) * NH + h) * ND + dt * 16 + n] = acc[r];
    }
}

extern "C" void kernel_launch(void* const* d_in, const int* in_sizes, int n_in,
                              void* d_out, int out_size, void* d_ws, size_t ws_size,
                              hipStream_t stream) {
    const float* queries = (const float*)d_in[0];
    const float* keys    = (const float*)d_in[1];
    const float* values  = (const float*)d_in[2];
    const float* wq      = (const float*)d_in[3];
    const float* bq      = (const float*)d_in[4];
    const float* wk      = (const float*)d_in[5];
    const float* bk      = (const float*)d_in[6];

    float* ctx  = (float*)d_out;                       // [B,L,H,64]
    float* attn = ctx + (size_t)NB * NL * NH * ND;     // [B,H,L,S]

    char* ws = (char*)d_ws;
    f16* QA = (f16*)(ws);                              // 4 MB  [B,H,L,32]
    f16* QB = (f16*)(ws + (4u  << 20));                // 4 MB
    f16* KS = (f16*)(ws + (8u  << 20));                // 4 MB  [B,H,S,32]
    f16* Vh = (f16*)(ws + (12u << 20));                // 8 MB  [B,H,S,64]

    const int rowsTotal = NB * NH * NL;                // 65536 token rows each side

    qka_encode<<<rowsTotal / 8, 256, 0, stream>>>(queries, wq, bq, QA, QB, 1);
    qka_encode<<<rowsTotal / 8, 256, 0, stream>>>(keys,    wk, bk, KS, QB, 0);
    qka_convert_v<<<(NB * NH * NL * ND) / 256, 256, 0, stream>>>(values, Vh);
    qka_scores<<<(NB * NH * (NL / 16)) / 8, 256, 0, stream>>>(QA, QB, KS, attn);
    qka_context<<<NB * NH * (NL / 32), 256, 0, stream>>>(attn, Vh, ctx);
    (void)in_sizes; (void)n_in; (void)out_size; (void)ws_size;
}
